// LSTM_42494406426876
// MI455X (gfx1250) — matrix operand order repaired and checked by reading, not compiled, not run on hardware
//
#include <hip/hip_runtime.h>
#include <hip/hip_bf16.h>

typedef __bf16 bf16_t;
typedef __attribute__((ext_vector_type(16))) __bf16 v16bf;
typedef __attribute__((ext_vector_type(8)))  __bf16 v8bf;
typedef __attribute__((ext_vector_type(8)))  float  v8f;

constexpr int Bb = 64;
constexpr int Tt = 512;
constexpr int Dd = 1024;
constexpr int Hh = 1024;
constexpr int G4 = 4 * Hh;          // 4096
constexpr int BT = Bb * Tt;         // 32768
constexpr int KC = 256;             // K-chunk staged in LDS per iteration

#define WMMA_BF16(a, b, c) \
    __builtin_amdgcn_wmma_f32_16x16x32_bf16(false, (a), false, (b), (short)0, (c), false, false)

// ---------------------------------------------------------------------------
// Global-memory fragment loaders (used by the one-shot input GEMM).
// A: 16x32 bf16 of row-major MxK. Contiguous layout (matches B):
//    lanes 0-15: M=lane, K=0..15; lanes 16-31: M=lane-16, K=16..31
// B: 32x16 where B[k][n] = W[n][k], W row-major: lane holds 16 contiguous K.
// ---------------------------------------------------------------------------
__device__ inline v16bf load_a_frag_g(const bf16_t* p, int ld, int m0, int k0) {
    int lane = threadIdx.x & 31;
    int row  = m0 + (lane & 15);
    int half = (lane >> 4) << 4;
    const bf16_t* base = p + (size_t)row * ld + k0 + half;
    v8bf lo = *(const v8bf*)(base);
    v8bf hi = *(const v8bf*)(base + 8);
    v16bf r;
#pragma unroll
    for (int i = 0; i < 8; ++i) { r[i] = lo[i]; r[8 + i] = hi[i]; }
    return r;
}

__device__ inline v16bf load_b_frag_g(const bf16_t* w, int ld, int n0, int k0) {
    int lane = threadIdx.x & 31;
    int col  = n0 + (lane & 15);
    int koff = k0 + ((lane >> 4) << 4);
    return *(const v16bf*)(w + (size_t)col * ld + koff);
}

// Same fragment layouts, sourced from LDS tiles with leading dimension KC.
__device__ inline v16bf load_a_frag_lds(const bf16_t* p, int row0, int k0) {
    int lane = threadIdx.x & 31;
    const bf16_t* base = p + (row0 + (lane & 15)) * KC + k0 + (((threadIdx.x & 31) >> 4) << 4);
    v8bf lo = *(const v8bf*)(base);
    v8bf hi = *(const v8bf*)(base + 8);
    v16bf r;
#pragma unroll
    for (int i = 0; i < 8; ++i) { r[i] = lo[i]; r[8 + i] = hi[i]; }
    return r;
}

__device__ inline v16bf load_b_frag_lds(const bf16_t* p, int row0, int k0) {
    int lane = threadIdx.x & 31;
    int row  = row0 + (lane & 15);
    int koff = k0 + ((lane >> 4) << 4);
    return *(const v16bf*)(p + row * KC + koff);
}

// Async copy of one 16-byte packet global -> LDS (CDNA5 path, ASYNCcnt).
__device__ inline void async_cp16(const bf16_t* gsrc, bf16_t* ldst) {
    unsigned lds_off = (unsigned)(size_t)ldst;   // AS3 offset = low 32 bits
    asm volatile("global_load_async_to_lds_b128 %0, %1, off"
                 :: "v"(lds_off), "v"(gsrc) : "memory");
}
__device__ inline void async_wait0() {
    asm volatile("s_wait_asynccnt 0x0" ::: "memory");
}

// f32 -> bf16 conversion, 4 elements/thread
__global__ void f32_to_bf16x4(const float* __restrict__ s, bf16_t* __restrict__ d, int n4) {
    int i = blockIdx.x * blockDim.x + threadIdx.x;
    if (i < n4) {
        const float4 v = ((const float4*)s)[i];
        bf16_t* o = d + 4 * (size_t)i;
        o[0] = (bf16_t)v.x; o[1] = (bf16_t)v.y; o[2] = (bf16_t)v.z; o[3] = (bf16_t)v.w;
    }
}

// zero h ping-pong (bf16, 2*B*H) and c (f32, B*H)
__global__ void init_state(bf16_t* h, float* c, int n) {
    int i = blockIdx.x * blockDim.x + threadIdx.x;
    if (i < n) { h[i] = (bf16_t)0.0f; h[i + n] = (bf16_t)0.0f; c[i] = 0.0f; }
}

// ---------------------------------------------------------------------------
// Phase 1: xg[m][g] = sum_k feat[m][k]*W_ih[g][k] + (b_ih[g]+b_hh[g])
// One wave -> 16x64 output strip (4 N-tiles, A-fragment reuse). One-shot,
// bandwidth-friendly; compiler already pipelines loads against WMMA.
// ---------------------------------------------------------------------------
__global__ void __launch_bounds__(256)
gemm_xgates(const bf16_t* __restrict__ feat, const bf16_t* __restrict__ wih,
            const float* __restrict__ b_ih, const float* __restrict__ b_hh,
            float* __restrict__ xg) {
    int wave = blockIdx.x * (blockDim.x >> 5) + (threadIdx.x >> 5);
    int lane = threadIdx.x & 31;
    const int NG = G4 / 64;
    int mt = wave / NG;
    int ng = wave % NG;
    int m0 = mt * 16;
    int n0 = ng * 64;

    v8f a0 = {}, a1 = {}, a2 = {}, a3 = {};
    for (int k0 = 0; k0 < Dd; k0 += 32) {
        v16bf a  = load_a_frag_g(feat, Dd, m0, k0);
        v16bf b0 = load_b_frag_g(wih, Dd, n0 +  0, k0);
        v16bf b1 = load_b_frag_g(wih, Dd, n0 + 16, k0);
        v16bf b2 = load_b_frag_g(wih, Dd, n0 + 32, k0);
        v16bf b3 = load_b_frag_g(wih, Dd, n0 + 48, k0);
        a0 = WMMA_BF16(a, b0, a0);
        a1 = WMMA_BF16(a, b1, a1);
        a2 = WMMA_BF16(a, b2, a2);
        a3 = WMMA_BF16(a, b3, a3);
    }

    int mrow = m0 + ((lane >> 4) << 3);
    int nl   = lane & 15;
#pragma unroll
    for (int tsel = 0; tsel < 4; ++tsel) {
        const v8f& acc = (tsel == 0) ? a0 : (tsel == 1) ? a1 : (tsel == 2) ? a2 : a3;
        int n = n0 + 16 * tsel + nl;
        float bias = b_ih[n] + b_hh[n];
#pragma unroll
        for (int r = 0; r < 8; ++r)
            xg[(size_t)(mrow + r) * G4 + n] = acc[r] + bias;
    }
}

__device__ inline float sigmoidf_(float x) { return 1.0f / (1.0f + __expf(-x)); }

// ---------------------------------------------------------------------------
// Phase 2: one recurrent step, LDS-staged.
// Block = 8 waves: wave w -> mt = w&3 (batch tile), ntl = w>>2 (hidden tile),
// block covers batch 0..63 x 32 hidden columns x all 4 gates.
// Per 256-K chunk: async-stage A (64xKC of h, 32KB) and B (4 gates x 32
// W_hh rows x KC, 64KB) into LDS, then WMMA out of LDS. Every operand
// byte enters the block exactly once.
// ---------------------------------------------------------------------------
__global__ void __launch_bounds__(256)
lstm_step(const bf16_t* __restrict__ h_in, bf16_t* __restrict__ h_out,
          const bf16_t* __restrict__ whh, const float* __restrict__ xg,
          float* __restrict__ c, const int* __restrict__ seq_len,
          float* __restrict__ out, int t) {
    __shared__ bf16_t Alds[64 * KC];    // 32 KB: h rows 0..63
    __shared__ bf16_t Blds[128 * KC];   // 64 KB: rows = gate*32 + j (j = hidden col in pair)

    int tid  = threadIdx.x;
    int lane = tid & 31;
    int w    = tid >> 5;                // 0..7
    int mt   = w & 3;                   // batch tile
    int ntl  = w >> 2;                  // 0 or 1
    int nbase = blockIdx.x * 32;        // first hidden column of this block
    int m0 = mt * 16;
    int n0 = nbase + ntl * 16;

    v8f ai = {}, af = {}, ag = {}, ao = {};

    for (int kc = 0; kc < Hh; kc += KC) {
        // ---- stage A: 64 rows x KC cols = 2048 16B packets, 8 per thread ----
#pragma unroll
        for (int u = tid; u < 64 * (KC / 8); u += 256) {
            int row = u >> 5;           // KC/8 = 32 packets per row
            int cu  = u & 31;
            async_cp16(h_in + (size_t)row * Hh + kc + cu * 8,
                       Alds + row * KC + cu * 8);
        }
        // ---- stage B: 128 rows x KC cols = 4096 packets, 16 per thread ----
#pragma unroll
        for (int u = tid; u < 128 * (KC / 8); u += 256) {
            int lr = u >> 5;            // 0..127: gate*32 + j
            int cu = u & 31;
            int gate = lr >> 5;
            int j    = lr & 31;
            const bf16_t* gsrc = whh + (size_t)(gate * Hh + nbase + j) * Hh + kc + cu * 8;
            async_cp16(gsrc, Blds + lr * KC + cu * 8);
        }
        async_wait0();
        __syncthreads();

        // ---- compute: 8 K-iterations of 32, 4 gates each ----
#pragma unroll
        for (int k0 = 0; k0 < KC; k0 += 32) {
            v16bf a  = load_a_frag_lds(Alds, m0, k0);
            v16bf bi = load_b_frag_lds(Blds, 0 * 32 + ntl * 16, k0);
            v16bf bf = load_b_frag_lds(Blds, 1 * 32 + ntl * 16, k0);
            v16bf bg = load_b_frag_lds(Blds, 2 * 32 + ntl * 16, k0);
            v16bf bo = load_b_frag_lds(Blds, 3 * 32 + ntl * 16, k0);
            ai = WMMA_BF16(a, bi, ai);
            af = WMMA_BF16(a, bf, af);
            ag = WMMA_BF16(a, bg, ag);
            ao = WMMA_BF16(a, bo, ao);
        }
        __syncthreads();                // LDS reuse across chunks
    }

    // ---- fused LSTM cell epilogue ----
    int n    = n0 + (lane & 15);
    int mrow = m0 + ((lane >> 4) << 3);
#pragma unroll
    for (int r = 0; r < 8; ++r) {
        int b = mrow + r;
        size_t xbase = ((size_t)b * Tt + t) * G4;
        float gi = sigmoidf_(xg[xbase + 0 * Hh + n] + ai[r]);
        float gf = sigmoidf_(xg[xbase + 1 * Hh + n] + af[r]);
        float gg = tanhf    (xg[xbase + 2 * Hh + n] + ag[r]);
        float go = sigmoidf_(xg[xbase + 3 * Hh + n] + ao[r]);
        size_t ci = (size_t)b * Hh + n;
        size_t oi = ((size_t)b * Tt + t) * Hh + n;
        if (t < seq_len[b]) {
            float cnew = gf * c[ci] + gi * gg;
            float hnew = go * tanhf(cnew);
            c[ci]     = cnew;
            h_out[ci] = (bf16_t)hnew;
            out[oi]   = hnew;
        } else {
            h_out[ci] = h_in[ci];       // carry h forward
            out[oi]   = 0.0f;           // reference zeroes masked outputs
        }
    }
}

extern "C" void kernel_launch(void* const* d_in, const int* in_sizes, int n_in,
                              void* d_out, int out_size, void* d_ws, size_t ws_size,
                              hipStream_t stream) {
    const float* feature = (const float*)d_in[0];   // (B,T,D)
    const float* W_ih    = (const float*)d_in[1];   // (4H,D)
    const float* W_hh    = (const float*)d_in[2];   // (4H,H)
    const float* b_ih    = (const float*)d_in[3];   // (4H)
    const float* b_hh    = (const float*)d_in[4];   // (4H)
    const int*   seq_len = (const int*)d_in[5];     // (B)
    float*       out     = (float*)d_out;           // (B,T,H)

    char* ws = (char*)d_ws;
    size_t off = 0;
    bf16_t* feat_bf = (bf16_t*)(ws + off); off += (size_t)BT * Dd * sizeof(bf16_t);
    bf16_t* wih_bf  = (bf16_t*)(ws + off); off += (size_t)G4 * Dd * sizeof(bf16_t);
    bf16_t* whh_bf  = (bf16_t*)(ws + off); off += (size_t)G4 * Hh * sizeof(bf16_t);
    float*  xg      = (float*) (ws + off); off += (size_t)BT * G4 * sizeof(float);
    bf16_t* hbuf    = (bf16_t*)(ws + off); off += (size_t)2 * Bb * Hh * sizeof(bf16_t);
    float*  cbuf    = (float*) (ws + off); off += (size_t)Bb * Hh * sizeof(float);

    // 1) bf16 conversions
    {
        int n4 = BT * Dd / 4;
        f32_to_bf16x4<<<(n4 + 255) / 256, 256, 0, stream>>>(feature, feat_bf, n4);
        n4 = G4 * Dd / 4;
        f32_to_bf16x4<<<(n4 + 255) / 256, 256, 0, stream>>>(W_ih, wih_bf, n4);
        n4 = G4 * Hh / 4;
        f32_to_bf16x4<<<(n4 + 255) / 256, 256, 0, stream>>>(W_hh, whh_bf, n4);
    }

    // 2) zero h ping-pong + c
    init_state<<<(Bb * Hh + 255) / 256, 256, 0, stream>>>(hbuf, cbuf, Bb * Hh);

    // 3) input-projection GEMM: 2048 M-tiles x 64 N-strips = 131072 waves
    {
        int waves = (BT / 16) * (G4 / 64);
        gemm_xgates<<<waves / 8, 256, 0, stream>>>(feat_bf, wih_bf, b_ih, b_hh, xg);
    }

    // 4) recurrent scan: 512 dependent steps, 32 blocks each
    for (int t = 0; t < Tt; ++t) {
        const bf16_t* h_in  = hbuf + (size_t)(t & 1) * Bb * Hh;
        bf16_t*       h_out = hbuf + (size_t)((t + 1) & 1) * Bb * Hh;
        lstm_step<<<32, 256, 0, stream>>>(h_in, h_out, whh_bf, xg, cbuf, seq_len, out, t);
    }
}
